// MLPDecoder_19636590477699
// MI455X (gfx1250) — matrix operand order repaired and checked by reading, not compiled, not run on hardware
//
#include <hip/hip_runtime.h>

typedef __attribute__((ext_vector_type(16))) _Float16 v16h;
typedef __attribute__((ext_vector_type(8)))  _Float16 v8h;
typedef __attribute__((ext_vector_type(8)))  float    v8f;
typedef __attribute__((ext_vector_type(4)))  float    f4;

#define DIM   128
#define HID   128
#define NT    8                        // 8 N-tiles of 16 -> HID = 128
#define KCH   12                       // 12 K-chunks of 32 -> 3*DIM = 384
#define W1_LDS_HALFS (NT * KCH * 32 * 16)   // 49152 halves = 96 KB

// LDS layout: [nt][kc][lane][16 halves].  kc stride = 32 lanes * 32 B = 1024 B,
// so the three K-chunks (c4, 4+c4, 8+c4) of one nt are at +0 / +4096 / +8192
// from one base address -> all reachable with 16-bit DS immediate offsets.

__device__ __forceinline__ v16h cat8(v8h lo, v8h hi) {
  return __builtin_shufflevector(lo, hi, 0, 1, 2, 3, 4, 5, 6, 7,
                                 8, 9, 10, 11, 12, 13, 14, 15);
}

// Load the three B-fragments (zu / zv / |zu-zv| rows of W1) for one N-tile.
// Inline asm ds_load_b128: guaranteed DS path, cannot be hoisted/spilled.
__device__ __forceinline__ void lds_b3(unsigned addr, v16h& bu, v16h& bv, v16h& ba) {
  v8h p0, p1, p2, p3, p4, p5;
  asm volatile(
      "ds_load_b128 %0, %6\n\t"
      "ds_load_b128 %1, %6 offset:16\n\t"
      "ds_load_b128 %2, %6 offset:4096\n\t"
      "ds_load_b128 %3, %6 offset:4112\n\t"
      "ds_load_b128 %4, %6 offset:8192\n\t"
      "ds_load_b128 %5, %6 offset:8208\n\t"
      "s_wait_dscnt 0x0"
      : "=v"(p0), "=v"(p1), "=v"(p2), "=v"(p3), "=v"(p4), "=v"(p5)
      : "v"(addr));
  bu = cat8(p0, p1);
  bv = cat8(p2, p3);
  ba = cat8(p4, p5);
}

__global__ __launch_bounds__(256)
void mlp_decoder_wmma(const float* __restrict__ z,
                      const long long* __restrict__ edge,   // [2, E] int64
                      const float* __restrict__ W1,         // [384, 128]
                      const float* __restrict__ b1,         // [128]
                      const float* __restrict__ W2,         // [128, 1]
                      const float* __restrict__ b2,         // [1]
                      float* __restrict__ out,              // [E]
                      int E, int ntiles) {
  extern __shared__ _Float16 lw1[];

  const int tid   = threadIdx.x;
  const int lane  = tid & 31;
  const int wid   = tid >> 5;
  const int mrow  = lane & 15;     // edge row within the 16-edge tile
  const int halfv = lane >> 4;     // 0: lanes 0-15, 1: lanes 16-31
  const int kbase = halfv << 4;    // A-fragment K sub-base: 0 or 16

  // ---- stage W1 (f32 -> f16) into LDS in per-lane WMMA B-fragment order ----
  // B (32x16 per chunk): lane l holds column n = l%16; K half = 16*(l/16);
  // element i within the fragment maps to k = khalf + i.
  for (int f = tid; f < NT * KCH * 32; f += (int)blockDim.x) {
    const int l   = f & 31;
    const int rem = f >> 5;
    const int kc  = rem % KCH;
    const int nt  = rem / KCH;
    const int n   = nt * 16 + (l & 15);
    const int kb  = kc * 32 + ((l >> 4) << 4);
    _Float16* dst = lw1 + ((size_t)f << 4);
#pragma unroll
    for (int i = 0; i < 16; ++i)
      dst[i] = (_Float16)W1[(kb + i) * HID + n];
  }
  __syncthreads();

  // per-lane column constants (column n = nt*16 + mrow), hoisted out of tile loop
  float b1v[NT], w2v[NT];
#pragma unroll
  for (int nt = 0; nt < NT; ++nt) {
    b1v[nt] = b1[nt * 16 + mrow];
    w2v[nt] = W2[nt * 16 + mrow];
  }
  const float b2s = b2[0];

  // DS byte address of this lane's fragment slot 0 (generic-ptr low 32 bits
  // are the LDS offset per the flat->LDS aperture mapping).
  const unsigned lwbase = (unsigned)(uintptr_t)lw1 + ((unsigned)lane << 5);

  const int wave   = blockIdx.x * ((int)blockDim.x >> 5) + wid;
  const int nwaves = gridDim.x * ((int)blockDim.x >> 5);

  for (int tile = wave; tile < ntiles; tile += nwaves) {
    const int tb = tile * 16;
    const int m  = tb + mrow;
    const long long ui = (m < E) ? edge[m]     : 0;
    const long long vi = (m < E) ? edge[E + m] : 0;
    const float* zu = z + ui * DIM;
    const float* zv = z + vi * DIM;

    // C/D accumulators: VGPR r -> M = 8*halfv + r, N = nt*16 + mrow. Init with b1.
    v8f acc[NT];
#pragma unroll
    for (int nt = 0; nt < NT; ++nt) {
      v8f c;
#pragma unroll
      for (int r = 0; r < 8; ++r) c[r] = b1v[nt];
      acc[nt] = c;
    }

#pragma unroll
    for (int c4 = 0; c4 < 4; ++c4) {
      const int j0 = c4 * 32 + kbase;
      // This lane's 16 A-slots for the chunk: feats j0..j0+15.
      f4 u0 = *(const f4*)(zu + j0);
      f4 u1 = *(const f4*)(zu + j0 + 4);
      f4 u2 = *(const f4*)(zu + j0 + 8);
      f4 u3 = *(const f4*)(zu + j0 + 12);
      f4 v0 = *(const f4*)(zv + j0);
      f4 v1 = *(const f4*)(zv + j0 + 4);
      f4 v2 = *(const f4*)(zv + j0 + 8);
      f4 v3 = *(const f4*)(zv + j0 + 12);

      v16h au, av, aa;
#pragma unroll
      for (int i = 0; i < 4; ++i) {
        au[i]      = (_Float16)u0[i];  au[4 + i]  = (_Float16)u1[i];
        au[8 + i]  = (_Float16)u2[i];  au[12 + i] = (_Float16)u3[i];
        av[i]      = (_Float16)v0[i];  av[4 + i]  = (_Float16)v1[i];
        av[8 + i]  = (_Float16)v2[i];  av[12 + i] = (_Float16)v3[i];
        aa[i]      = (_Float16)fabsf(u0[i] - v0[i]);   // |zu-zv| in fp32, then cvt
        aa[4 + i]  = (_Float16)fabsf(u1[i] - v1[i]);
        aa[8 + i]  = (_Float16)fabsf(u2[i] - v2[i]);
        aa[12 + i] = (_Float16)fabsf(u3[i] - v3[i]);
      }

#pragma unroll
      for (int nt = 0; nt < NT; ++nt) {
        // base of (nt, c4); +4096 -> kc=4+c4 (zv), +8192 -> kc=8+c4 (|zu-zv|)
        const unsigned addr = lwbase + (unsigned)((nt * KCH + c4) << 10);
        v16h bu, bv, ba;
        lds_b3(addr, bu, bv, ba);
        acc[nt] = __builtin_amdgcn_wmma_f32_16x16x32_f16(
            false, au, false, bu, (short)0, acc[nt], false, false);
        acc[nt] = __builtin_amdgcn_wmma_f32_16x16x32_f16(
            false, av, false, bv, (short)0, acc[nt], false, false);
        acc[nt] = __builtin_amdgcn_wmma_f32_16x16x32_f16(
            false, aa, false, ba, (short)0, acc[nt], false, false);
      }
    }

    // Layer 2: ReLU + dot with W2 column slice; each 16-lane half owns the same
    // 8 rows across all 128 columns -> xor-shuffle reduce within the half.
    float part[8];
#pragma unroll
    for (int r = 0; r < 8; ++r) part[r] = 0.f;
#pragma unroll
    for (int nt = 0; nt < NT; ++nt) {
#pragma unroll
      for (int r = 0; r < 8; ++r)
        part[r] += fmaxf(acc[nt][r], 0.f) * w2v[nt];
    }
#pragma unroll
    for (int off = 1; off < 16; off <<= 1) {
#pragma unroll
      for (int r = 0; r < 8; ++r)
        part[r] += __shfl_xor(part[r], off, 32);
    }
    if (mrow == 0) {
      const int mb = tb + halfv * 8;
#pragma unroll
      for (int r = 0; r < 8; ++r)
        if (mb + r < E) out[mb + r] = part[r] + b2s;
    }
  }
}

extern "C" void kernel_launch(void* const* d_in, const int* in_sizes, int n_in,
                              void* d_out, int out_size, void* d_ws, size_t ws_size,
                              hipStream_t stream) {
  const float*     z    = (const float*)d_in[0];
  const long long* edge = (const long long*)d_in[1];
  const float*     W1   = (const float*)d_in[2];
  const float*     b1   = (const float*)d_in[3];
  const float*     W2   = (const float*)d_in[4];
  const float*     b2   = (const float*)d_in[5];
  float*           out  = (float*)d_out;

  const int E      = in_sizes[1] / 2;        // edge_index is [2, E]
  const int ntiles = (E + 15) / 16;

  const int wavesPerBlock = 8;               // 256 threads = 8 wave32
  int blocks = (ntiles + wavesPerBlock - 1) / wavesPerBlock;
  if (blocks > 2048) blocks = 2048;          // persistent waves grid-stride tiles

  const size_t shmem = (size_t)W1_LDS_HALFS * sizeof(_Float16);  // 96 KB
  (void)hipFuncSetAttribute((const void*)mlp_decoder_wmma,
                            hipFuncAttributeMaxDynamicSharedMemorySize,
                            (int)shmem);

  mlp_decoder_wmma<<<blocks, 256, shmem, stream>>>(z, edge, W1, b1, W2, b2,
                                                   out, E, ntiles);
}
